// GPT_2980707304222
// MI455X (gfx1250) — matrix operand order repaired and checked by reading, not compiled, not run on hardware
//
#include <hip/hip_runtime.h>

#define BB   2
#define TT   1024
#define DD   768
#define HH   12
#define HDD  64
#define KTK  4
#define MTR  8
#define NSY  512
#define HIDN 16
#define BT   (BB*TT)

typedef __attribute__((ext_vector_type(16))) _Float16 v16h;
typedef __attribute__((ext_vector_type(4)))  _Float16 v4h;
typedef __attribute__((ext_vector_type(8)))  float    v8f;

__device__ __forceinline__ float sigm_(float x) { return 1.f / (1.f + __expf(-x)); }

// ---------------------------------------------------------------------------
// WMMA GEMM:  C(2048xN) = act( concat(A1|A2)(2048xK) * B(KxN) )
// All K/N dims in this net are multiples of 4, so float4 chunks never straddle
// the concat boundary or K/N edges and are 16B aligned. Edge chunks use
// CLAMPED UNCONDITIONAL loads + post-load zero select (no exec-mask regions,
// all tile loads stay in flight together). LDS is double-buffered and the
// next tile's global loads are issued before the current tile's WMMAs.
// Block: 256 thr = 8 waves; block tile 64(M) x 128(N); wave tile 32x32
// (2x2 WMMA tiles); K-step 32.
// ---------------------------------------------------------------------------
__global__ __launch_bounds__(256) void gemm_wmma(
    const float* __restrict__ A1, const float* __restrict__ A2,
    int K1, int K2, const float* __restrict__ Bm,
    float* __restrict__ C, int N, int act)
{
    const int Ktot = K1 + K2;
    __shared__ __align__(16) _Float16 As[2][64][32];
    __shared__ __align__(16) _Float16 Bs[2][128][32];   // [n][k] for contiguous k-pairs

    const int tid  = threadIdx.x;
    const int wave = tid >> 5, lane = tid & 31;
    const int wm = wave >> 2, wn = wave & 3;
    const int mbase = blockIdx.y * 64, nbase = blockIdx.x * 128;
    const int mrow = lane & 15, khalf = lane >> 4;

    // fixed per-thread staging coordinates
    const int rA0 = tid >> 3,          cA0 = (tid & 7) * 4;          // A chunk j=0
    const int rA1 = (tid + 256) >> 3,  cA1 = ((tid + 256) & 7) * 4;  // A chunk j=1

    float4 ra[2], rb[4];

    auto load_tile = [&](int k0) {
        {   // A chunks
            const int rr[2] = { rA0, rA1 };
            const int cc[2] = { cA0, cA1 };
            #pragma unroll
            for (int j = 0; j < 2; ++j) {
                const int gcol = k0 + cc[j];
                const int gc = (gcol < Ktot - 4) ? gcol : (Ktot - 4);
                const float* p = (gc < K1)
                    ? A1 + (size_t)(mbase + rr[j]) * K1 + gc
                    : A2 + (size_t)(mbase + rr[j]) * K2 + (gc - K1);
                float4 v = *(const float4*)p;
                if (gcol >= Ktot) v = make_float4(0.f, 0.f, 0.f, 0.f);
                ra[j] = v;
            }
        }
        #pragma unroll
        for (int j = 0; j < 4; ++j) {   // B chunks
            const int e4 = tid + j * 256;
            const int kk = e4 >> 5, nn4 = (e4 & 31) * 4;
            const int gk = k0 + kk, gn = nbase + nn4;
            const int gkc = (gk < Ktot - 1) ? gk : (Ktot - 1);
            const int gnc = (gn < N - 4) ? gn : (N - 4);
            float4 v = *(const float4*)(Bm + (size_t)gkc * N + gnc);
            if (gk >= Ktot || gn >= N) v = make_float4(0.f, 0.f, 0.f, 0.f);
            rb[j] = v;
        }
    };
    auto store_tile = [&](int buf) {
        {
            const int rr[2] = { rA0, rA1 };
            const int cc[2] = { cA0, cA1 };
            #pragma unroll
            for (int j = 0; j < 2; ++j) {
                v4h hv;
                hv[0] = (_Float16)ra[j].x; hv[1] = (_Float16)ra[j].y;
                hv[2] = (_Float16)ra[j].z; hv[3] = (_Float16)ra[j].w;
                *(v4h*)&As[buf][rr[j]][cc[j]] = hv;
            }
        }
        #pragma unroll
        for (int j = 0; j < 4; ++j) {
            const int e4 = tid + j * 256;
            const int kk = e4 >> 5, nn4 = (e4 & 31) * 4;
            Bs[buf][nn4 + 0][kk] = (_Float16)rb[j].x;
            Bs[buf][nn4 + 1][kk] = (_Float16)rb[j].y;
            Bs[buf][nn4 + 2][kk] = (_Float16)rb[j].z;
            Bs[buf][nn4 + 3][kk] = (_Float16)rb[j].w;
        }
    };

    v8f a00 = {}, a01 = {}, a10 = {}, a11 = {};
    const int nk = (Ktot + 31) >> 5;
    load_tile(0);
    int buf = 0;
    for (int kt = 0; kt < nk; ++kt) {
        store_tile(buf);
        __syncthreads();
        if (kt + 1 < nk) load_tile((kt + 1) * 32);   // next tile in flight

        v16h af0, af1, bf0, bf1;
        #pragma unroll
        for (int v = 0; v < 8; ++v) {
            const int ka = ((v >= 4) ? 16 : 0) + khalf * 8 + (v & 3) * 2; // A packing
            const int kb = ((v >= 4) ? 16 : 0) + khalf * 8 + (v & 3) * 2; // B packing
            af0[2 * v]     = As[buf][wm * 32 + mrow][ka];
            af0[2 * v + 1] = As[buf][wm * 32 + mrow][ka + 1];
            af1[2 * v]     = As[buf][wm * 32 + 16 + mrow][ka];
            af1[2 * v + 1] = As[buf][wm * 32 + 16 + mrow][ka + 1];
            bf0[2 * v]     = Bs[buf][wn * 32 + mrow][kb];
            bf0[2 * v + 1] = Bs[buf][wn * 32 + mrow][kb + 1];
            bf1[2 * v]     = Bs[buf][wn * 32 + 16 + mrow][kb];
            bf1[2 * v + 1] = Bs[buf][wn * 32 + 16 + mrow][kb + 1];
        }
        a00 = __builtin_amdgcn_wmma_f32_16x16x32_f16(false, af0, false, bf0, (short)0, a00, false, false);
        a01 = __builtin_amdgcn_wmma_f32_16x16x32_f16(false, af0, false, bf1, (short)0, a01, false, false);
        a10 = __builtin_amdgcn_wmma_f32_16x16x32_f16(false, af1, false, bf0, (short)0, a10, false, false);
        a11 = __builtin_amdgcn_wmma_f32_16x16x32_f16(false, af1, false, bf1, (short)0, a11, false, false);
        buf ^= 1;
    }
    #pragma unroll
    for (int i = 0; i < 8; ++i) {
        const int r0 = mbase + wm * 32 + i + 8 * khalf;
        const int r1 = r0 + 16;
        const int c0 = nbase + wn * 32 + mrow;
        const int c1 = c0 + 16;
        if (c0 < N) {
            float u = a00[i], w = a10[i];
            if (act == 1) { u *= sigm_(u); w *= sigm_(w); }
            C[(size_t)r0 * N + c0] = u;
            C[(size_t)r1 * N + c0] = w;
        }
        if (c1 < N) {
            float u = a01[i], w = a11[i];
            if (act == 1) { u *= sigm_(u); w *= sigm_(w); }
            C[(size_t)r0 * N + c1] = u;
            C[(size_t)r1 * N + c1] = w;
        }
    }
}

// ---------------------------------------------------------------------------
// Flash attention (causal). One wave per 16-query tile, 8 tiles per block.
// Q already RMS-normed and pre-scaled by 1/sqrt(HD); K RMS-normed.
// ---------------------------------------------------------------------------
__global__ __launch_bounds__(256) void attn_kernel(
    const float* __restrict__ Q, const float* __restrict__ Kn,
    const float* __restrict__ V, float* __restrict__ O)
{
    const int bh = blockIdx.x;
    const int b = bh / HH, h = bh % HH;
    const int qblk = blockIdx.y;              // 0..7 (128 queries each)
    const int tid = threadIdx.x, wave = tid >> 5, lane = tid & 31;
    const int mrow = lane & 15, khalf = lane >> 4;

    __shared__ __align__(16) _Float16 Kt[32][64];
    __shared__ __align__(16) _Float16 Vt[32][64];
    __shared__ __align__(16) _Float16 Ps[8][16][32];

    const int qi = qblk * 128 + wave * 16;

    // load Q fragments (A-layout) with float2 pair loads
    v16h qf0, qf1;
    {
        const float* p = Q + (size_t)(b * TT + qi + mrow) * DD + h * HDD;
        #pragma unroll
        for (int v = 0; v < 8; ++v) {
            const int ka = ((v >= 4) ? 16 : 0) + khalf * 8 + (v & 3) * 2;
            const float2 q0 = *(const float2*)(p + ka);
            const float2 q1 = *(const float2*)(p + 32 + ka);
            qf0[2 * v] = (_Float16)q0.x; qf0[2 * v + 1] = (_Float16)q0.y;
            qf1[2 * v] = (_Float16)q1.x; qf1[2 * v + 1] = (_Float16)q1.y;
        }
    }

    v8f oacc[4] = {};
    float mx[8], lsum[8];
    #pragma unroll
    for (int i = 0; i < 8; ++i) { mx[i] = -1e30f; lsum[i] = 0.f; }

    const int nkb = (qblk + 1) * 4;           // 32-key blocks up to block's last q
    for (int kb = 0; kb < nkb; ++kb) {
        #pragma unroll
        for (int j = 0; j < 2; ++j) {         // 32 keys x 64 dims = 512 float4 chunks
            const int e4 = tid + j * 256;
            const int kk = e4 >> 4, dd4 = (e4 & 15) * 4;
            const size_t g = (size_t)(b * TT + kb * 32 + kk) * DD + h * HDD + dd4;
            const float4 kv = *(const float4*)(Kn + g);
            const float4 vv = *(const float4*)(V + g);
            v4h hk, hv;
            hk[0] = (_Float16)kv.x; hk[1] = (_Float16)kv.y;
            hk[2] = (_Float16)kv.z; hk[3] = (_Float16)kv.w;
            hv[0] = (_Float16)vv.x; hv[1] = (_Float16)vv.y;
            hv[2] = (_Float16)vv.z; hv[3] = (_Float16)vv.w;
            *(v4h*)&Kt[kk][dd4] = hk;
            *(v4h*)&Vt[kk][dd4] = hv;
        }
        __syncthreads();

        if (kb * 32 <= qi + 15) {
            v8f s[2];
            #pragma unroll
            for (int st = 0; st < 2; ++st) {  // S = Q * K^T, two 16-key tiles
                v16h bf0, bf1;
                #pragma unroll
                for (int v = 0; v < 8; ++v) {
                    const int kbd = ((v >= 4) ? 16 : 0) + khalf * 8 + (v & 3) * 2; // dim index
                    const int key = st * 16 + mrow;
                    bf0[2 * v]     = Kt[key][kbd];
                    bf0[2 * v + 1] = Kt[key][kbd + 1];
                    bf1[2 * v]     = Kt[key][32 + kbd];
                    bf1[2 * v + 1] = Kt[key][32 + kbd + 1];
                }
                v8f a = {};
                a = __builtin_amdgcn_wmma_f32_16x16x32_f16(false, qf0, false, bf0, (short)0, a, false, false);
                a = __builtin_amdgcn_wmma_f32_16x16x32_f16(false, qf1, false, bf1, (short)0, a, false, false);
                s[st] = a;
            }
            // causal mask + online softmax (row m = i + 8*khalf, col n = mrow)
            #pragma unroll
            for (int i = 0; i < 8; ++i) {
                const int qrow = qi + i + 8 * khalf;
                #pragma unroll
                for (int st = 0; st < 2; ++st) {
                    const int key = kb * 32 + st * 16 + mrow;
                    if (key > qrow) s[st][i] = -1e30f;
                }
                float m2 = fmaxf(s[0][i], s[1][i]);
                #pragma unroll
                for (int d = 1; d < 16; d <<= 1) m2 = fmaxf(m2, __shfl_xor(m2, d, 32));
                const float nm = fmaxf(mx[i], m2);
                const float corr = __expf(mx[i] - nm);
                s[0][i] = __expf(s[0][i] - nm);
                s[1][i] = __expf(s[1][i] - nm);
                float r2 = s[0][i] + s[1][i];
                #pragma unroll
                for (int d = 1; d < 16; d <<= 1) r2 += __shfl_xor(r2, d, 32);
                lsum[i] = lsum[i] * corr + r2;
                mx[i] = nm;
                #pragma unroll
                for (int nt = 0; nt < 4; ++nt) oacc[nt][i] *= corr;
            }
            // stage P (C-layout -> LDS) and re-load as A-fragment
            #pragma unroll
            for (int st = 0; st < 2; ++st)
                #pragma unroll
                for (int i = 0; i < 8; ++i)
                    Ps[wave][i + 8 * khalf][st * 16 + mrow] = (_Float16)s[st][i];
            v16h af;
            #pragma unroll
            for (int v = 0; v < 8; ++v) {
                const int ka = ((v >= 4) ? 16 : 0) + khalf * 8 + (v & 3) * 2;
                af[2 * v]     = Ps[wave][mrow][ka];
                af[2 * v + 1] = Ps[wave][mrow][ka + 1];
            }
            #pragma unroll
            for (int nt = 0; nt < 4; ++nt) {   // O += P * V
                v16h bv;
                #pragma unroll
                for (int v = 0; v < 8; ++v) {
                    const int kbd = ((v >= 4) ? 16 : 0) + khalf * 8 + (v & 3) * 2;
                    bv[2 * v]     = Vt[kbd][nt * 16 + mrow];
                    bv[2 * v + 1] = Vt[kbd + 1][nt * 16 + mrow];
                }
                oacc[nt] = __builtin_amdgcn_wmma_f32_16x16x32_f16(false, af, false, bv,
                                                                  (short)0, oacc[nt], false, false);
            }
        }
        __syncthreads();
    }
    #pragma unroll
    for (int nt = 0; nt < 4; ++nt)
        #pragma unroll
        for (int i = 0; i < 8; ++i) {
            const int q = qi + i + 8 * khalf;
            O[(size_t)(b * TT + q) * DD + h * HDD + nt * 16 + mrow] = oacc[nt][i] / lsum[i];
        }
}

// ---------------------------------------------------------------------------
// Elementwise / small kernels
// ---------------------------------------------------------------------------
__global__ void rms64_kernel(const float* __restrict__ in, float* __restrict__ out,
                             int rows, float outscale)
{
    const int wave = threadIdx.x >> 5, lane = threadIdx.x & 31;
    const int row = blockIdx.x * 8 + wave;
    if (row >= rows) return;
    const float2 ab = *(const float2*)(in + (size_t)row * 64 + lane * 2);
    float ss = ab.x * ab.x + ab.y * ab.y;
    #pragma unroll
    for (int d = 1; d < 32; d <<= 1) ss += __shfl_xor(ss, d, 32);
    const float sc = rsqrtf(ss / 64.f + 1.1920929e-07f) * outscale;
    float2 r; r.x = ab.x * sc; r.y = ab.y * sc;
    *(float2*)(out + (size_t)row * 64 + lane * 2) = r;
}

__global__ void ln_kernel(float* __restrict__ x, const float* __restrict__ g,
                          const float* __restrict__ bb, int W)
{
    float* p = x + (size_t)blockIdx.x * W;
    __shared__ float red[8];
    const int tid = threadIdx.x, lane = tid & 31, wave = tid >> 5;
    float s = 0.f;
    for (int c = tid; c < W; c += 256) s += p[c];
    #pragma unroll
    for (int d = 1; d < 32; d <<= 1) s += __shfl_xor(s, d, 32);
    if (lane == 0) red[wave] = s;
    __syncthreads();
    if (tid == 0) { float t = 0; for (int i = 0; i < 8; ++i) t += red[i]; red[0] = t / W; }
    __syncthreads();
    const float mean = red[0];
    __syncthreads();
    float v = 0.f;
    for (int c = tid; c < W; c += 256) { const float d0 = p[c] - mean; v += d0 * d0; }
    #pragma unroll
    for (int d = 1; d < 32; d <<= 1) v += __shfl_xor(v, d, 32);
    if (lane == 0) red[wave] = v;
    __syncthreads();
    if (tid == 0) { float t = 0; for (int i = 0; i < 8; ++i) t += red[i]; red[0] = rsqrtf(t / W + 1e-5f); }
    __syncthreads();
    const float inv = red[0];
    for (int c = tid; c < W; c += 256) p[c] = (p[c] - mean) * inv * g[c] + bb[c];
}

__global__ void init_state_kernel(float* st, const float* s0) {
    const int i = blockIdx.x * 256 + threadIdx.x;
    if (i < BT * DD) st[i] = s0[i % DD];
}
__global__ void init_trace_kernel(float* tr, const float* t0) {
    const int i = blockIdx.x * 256 + threadIdx.x;
    if (i < BT * DD * MTR) tr[i] = t0[((i >> 3) % DD) * MTR + (i & 7)];
}
__global__ void decay_kernel(const float* d, float* r, int n) {
    const int i = blockIdx.x * 256 + threadIdx.x;
    if (i < n) r[i] = __expf(-fminf(fmaxf(d[i], 0.f), 15.f));
}
__global__ void init_sync_kernel(const float* s0, const int* li, const int* ri,
                                 float* alpha, float* beta) {
    const int i = blockIdx.x * 256 + threadIdx.x;
    if (i >= BT * NSY) return;
    const int j = i % NSY;
    alpha[i] = s0[li[j]] * s0[ri[j]];
    beta[i] = 1.f;
}
__global__ void synch_kernel(const float* a, const float* b, float* o, int n) {
    const int i = blockIdx.x * 256 + threadIdx.x;
    if (i < n) o[i] = a[i] * rsqrtf(b[i]);
}
__global__ void addtick_kernel(const float* st, const float* tick, float* o) {
    const int i = blockIdx.x * 256 + threadIdx.x;
    if (i < BT * DD) o[i] = st[i] + tick[i % DD];
}
__global__ void newstate_trace_kernel(const float* st, const float* g4,
                                      float* trace, int slot) {
    const int i = blockIdx.x * 256 + threadIdx.x;
    if (i < BT * DD) trace[(size_t)i * MTR + slot] = st[i] + g4[i];
}
__global__ void nlm_kernel(const float* __restrict__ trace, const float* __restrict__ w1,
                           const float* __restrict__ b1, const float* __restrict__ w2,
                           const float* __restrict__ b2, float* __restrict__ stout, int c)
{
    const int idx = blockIdx.x * 256 + threadIdx.x;   // bt*DD + n
    if (idx >= BT * DD) return;
    const int n = idx % DD;
    float tr[MTR];
    const float* tp = trace + (size_t)idx * MTR;
    #pragma unroll
    for (int m = 0; m < MTR; ++m) tr[m] = tp[(c + m) & 7];
    float pre[2 * HIDN];
    #pragma unroll
    for (int o = 0; o < 2 * HIDN; ++o) {
        float s = b1[n * (2 * HIDN) + o];
        #pragma unroll
        for (int m = 0; m < MTR; ++m) s += tr[m] * w1[((size_t)m * (2 * HIDN) + o) * DD + n];
        pre[o] = s;
    }
    float h[HIDN];
    #pragma unroll
    for (int j = 0; j < HIDN; ++j) h[j] = pre[j] * sigm_(pre[HIDN + j]);
    float p0 = b2[n * 2 + 0], p1 = b2[n * 2 + 1];
    #pragma unroll
    for (int j = 0; j < HIDN; ++j) {
        p0 += h[j] * w2[((size_t)j * 2 + 0) * DD + n];
        p1 += h[j] * w2[((size_t)j * 2 + 1) * DD + n];
    }
    stout[idx] = p0 * sigm_(p1);
}
__global__ void sync_update_kernel(const float* __restrict__ st,
                                   const int* lio, const int* rio,
                                   const int* lia, const int* ria,
                                   const float* ro, const float* ra,
                                   float* ao, float* bo, float* aa, float* ba)
{
    const int i = blockIdx.x * 256 + threadIdx.x;
    if (i >= BT * NSY) return;
    const int j = i % NSY;
    const float* s = st + (size_t)(i / NSY) * DD;
    const float po = s[lio[j]] * s[rio[j]];
    const float pa = s[lia[j]] * s[ria[j]];
    ao[i] = ro[j] * ao[i] + po;
    bo[i] = ro[j] * bo[i] + 1.f;
    aa[i] = ra[j] * aa[i] + pa;
    ba[i] = ra[j] * ba[i] + 1.f;
}

// ---------------------------------------------------------------------------
extern "C" void kernel_launch(void* const* d_in, const int* in_sizes, int n_in,
                              void* d_out, int out_size, void* d_ws, size_t ws_size,
                              hipStream_t stream)
{
    (void)in_sizes; (void)n_in; (void)out_size; (void)ws_size;
    const float* x   = (const float*)d_in[0];
    const float* qW  = (const float*)d_in[1];
    const float* kW  = (const float*)d_in[2];
    const float* vW  = (const float*)d_in[3];
    const float* cW  = (const float*)d_in[4];
    const float* dw[4] = {(const float*)d_in[5], (const float*)d_in[6],
                          (const float*)d_in[7], (const float*)d_in[8]};
    const float* uw[4] = {(const float*)d_in[9], (const float*)d_in[10],
                          (const float*)d_in[11], (const float*)d_in[12]};
    const float* lnw[4] = {(const float*)d_in[13], (const float*)d_in[14],
                           (const float*)d_in[15], (const float*)d_in[16]};
    const float* lnb[4] = {(const float*)d_in[17], (const float*)d_in[18],
                           (const float*)d_in[19], (const float*)d_in[20]};
    const float* w1  = (const float*)d_in[21];
    const float* b1  = (const float*)d_in[22];
    const float* w2  = (const float*)d_in[23];
    const float* b2  = (const float*)d_in[24];
    const float* s0  = (const float*)d_in[25];
    const float* t0  = (const float*)d_in[26];
    const float* tke = (const float*)d_in[27];
    const float* dco = (const float*)d_in[28];
    const float* dca = (const float*)d_in[29];
    const int* lio = (const int*)d_in[30];
    const int* rio = (const int*)d_in[31];
    const int* lia = (const int*)d_in[32];
    const int* ria = (const int*)d_in[33];

    char* wsb = (char*)d_ws;
    size_t off = 0;
    auto alloc = [&](size_t nf) -> float* {
        float* p = (float*)(wsb + off); off += nf * sizeof(float); return p;
    };
    float* kpre  = alloc((size_t)BT * DD);   // attn_k (rms'd in place)
    float* vbuf  = alloc((size_t)BT * DD);
    float* qpre  = alloc((size_t)BT * DD);   // also reused as UNet g4
    float* obs   = alloc((size_t)BT * DD);
    float* state = alloc((size_t)BT * DD);
    float* stk   = alloc((size_t)BT * DD);
    float* sa    = alloc((size_t)BT * NSY);  // synch_act, reused as synch_out
    float* ao    = alloc((size_t)BT * NSY);
    float* bo    = alloc((size_t)BT * NSY);
    float* aa    = alloc((size_t)BT * NSY);
    float* ba    = alloc((size_t)BT * NSY);
    float* ro    = alloc(NSY);
    float* ra    = alloc(NSY);
    float* h1 = alloc((size_t)BT * 1156);
    float* h2 = alloc((size_t)BT * 776);
    float* h3 = alloc((size_t)BT * 396);
    float* h4 = alloc((size_t)BT * 16);
    float* g1 = alloc((size_t)BT * 396);
    float* g2 = alloc((size_t)BT * 776);
    float* g3 = alloc((size_t)BT * 1156);
    float* g4 = qpre;                        // q_pre dead after attention
    float* trace = alloc((size_t)BT * DD * MTR);
    float* outp = (float*)d_out;

    const dim3 blk(256);
    auto g1d = [](long n) { return dim3((unsigned)((n + 255) / 256)); };
    auto ggemm = [](int N) { return dim3((unsigned)((N + 127) / 128), BT / 64); };

    // ---- phase 0: K/V projections + norms, init state/trace/sync ----
    gemm_wmma<<<ggemm(DD), blk, 0, stream>>>(x, nullptr, DD, 0, kW, kpre, DD, 0);
    gemm_wmma<<<ggemm(DD), blk, 0, stream>>>(x, nullptr, DD, 0, vW, vbuf, DD, 0);
    rms64_kernel<<<dim3(BT * HH / 8), blk, 0, stream>>>(kpre, kpre, BT * HH, 1.f);
    init_state_kernel<<<g1d((long)BT * DD), blk, 0, stream>>>(state, s0);
    init_trace_kernel<<<g1d((long)BT * DD * MTR), blk, 0, stream>>>(trace, t0);
    decay_kernel<<<g1d(NSY), blk, 0, stream>>>(dco, ro, NSY);
    decay_kernel<<<g1d(NSY), blk, 0, stream>>>(dca, ra, NSY);
    init_sync_kernel<<<g1d((long)BT * NSY), blk, 0, stream>>>(s0, lio, rio, ao, bo);
    init_sync_kernel<<<g1d((long)BT * NSY), blk, 0, stream>>>(s0, lia, ria, aa, ba);

    // ---- K ticks ----
    for (int k = 0; k < KTK; ++k) {
        synch_kernel<<<g1d((long)BT * NSY), blk, 0, stream>>>(aa, ba, sa, BT * NSY);
        gemm_wmma<<<ggemm(DD), blk, 0, stream>>>(sa, nullptr, NSY, 0, qW, qpre, DD, 0);
        rms64_kernel<<<dim3(BT * HH / 8), blk, 0, stream>>>(qpre, qpre, BT * HH, 0.125f);
        attn_kernel<<<dim3(BB * HH, TT / 128), blk, 0, stream>>>(qpre, kpre, vbuf, obs);
        addtick_kernel<<<g1d((long)BT * DD), blk, 0, stream>>>(state, tke + (size_t)k * DD, stk);
        // UNet down
        gemm_wmma<<<ggemm(1156), blk, 0, stream>>>(obs, stk, DD, DD, dw[0], h1, 1156, 1);
        gemm_wmma<<<ggemm(776),  blk, 0, stream>>>(h1, nullptr, 1156, 0, dw[1], h2, 776, 1);
        gemm_wmma<<<ggemm(396),  blk, 0, stream>>>(h2, nullptr, 776, 0, dw[2], h3, 396, 1);
        gemm_wmma<<<ggemm(16),   blk, 0, stream>>>(h3, nullptr, 396, 0, dw[3], h4, 16, 1);
        // UNet up (skip concats via 2-source A)
        gemm_wmma<<<ggemm(396),  blk, 0, stream>>>(h4, nullptr, 16, 0, uw[0], g1, 396, 1);
        ln_kernel<<<dim3(BT), blk, 0, stream>>>(g1, lnw[0], lnb[0], 396);
        gemm_wmma<<<ggemm(776),  blk, 0, stream>>>(g1, h3, 396, 396, uw[1], g2, 776, 1);
        ln_kernel<<<dim3(BT), blk, 0, stream>>>(g2, lnw[1], lnb[1], 776);
        gemm_wmma<<<ggemm(1156), blk, 0, stream>>>(g2, h2, 776, 776, uw[2], g3, 1156, 1);
        ln_kernel<<<dim3(BT), blk, 0, stream>>>(g3, lnw[2], lnb[2], 1156);
        gemm_wmma<<<ggemm(DD),   blk, 0, stream>>>(g3, h1, 1156, 1156, uw[3], g4, DD, 1);
        ln_kernel<<<dim3(BT), blk, 0, stream>>>(g4, lnw[3], lnb[3], DD);
        // new_state -> trace ring slot, then NLM -> state
        newstate_trace_kernel<<<g1d((long)BT * DD), blk, 0, stream>>>(state, g4, trace, k & 7);
        nlm_kernel<<<g1d((long)BT * DD), blk, 0, stream>>>(trace, w1, b1, w2, b2, state, k + 1);
        sync_update_kernel<<<g1d((long)BT * NSY), blk, 0, stream>>>(
            state, lio, rio, lia, ria, ro, ra, ao, bo, aa, ba);
    }

    // ---- output projection ----
    synch_kernel<<<g1d((long)BT * NSY), blk, 0, stream>>>(ao, bo, sa, BT * NSY);
    gemm_wmma<<<ggemm(DD), blk, 0, stream>>>(sa, nullptr, NSY, 0, cW, outp, DD, 0);
}
